// AttentionHead_3753801417581
// MI455X (gfx1250) — compile-verified, matrix operand order checked by reading, not run on hardware
//
#include <hip/hip_runtime.h>

typedef _Float16 v16h  __attribute__((ext_vector_type(16)));
typedef float    v8f   __attribute__((ext_vector_type(8)));
typedef unsigned int u32v8 __attribute__((ext_vector_type(8)));

#define DM  512
#define DH  64
#define SEQ 4096

// K-striping pattern for 16-bit A/B operands (wave32, 16x16x32):
//   VGPR pair v (0..7), half = lane>>4 :  k = (v>>2)*16 + half*8 + (v&3)*2
__device__ __forceinline__ int kpat(int v, int half) {
    return ((v >> 2) << 4) + (half << 3) + ((v & 3) << 1);
}

__device__ __forceinline__ v8f wmma_f16(v16h a, v16h b, v8f c) {
    return __builtin_amdgcn_wmma_f32_16x16x32_f16(false, a, false, b,
                                                  (short)0, c, false, false);
}

// ---------------------------------------------------------------------------
// Kernel A: one projection (P=0: Q scaled by 1/sqrt(64); P=1: K; P=2: V
// stored transposed as (B, DH, SEQ)).  grid = BS/16, block = 128 (4 waves);
// wave w computes the 16x16 tile at columns [16w, 16w+16).
// ---------------------------------------------------------------------------
template <int P>
__global__ __launch_bounds__(128) void qkv_proj_kernel(
    const float* __restrict__ x, const float* __restrict__ W,
    const float* __restrict__ bias, _Float16* __restrict__ dst)
{
    const int lane = threadIdx.x & 31;
    const int wv   = threadIdx.x >> 5;
    const int half = lane >> 4;
    const int l15  = lane & 15;
    const int tile = blockIdx.x;

    const int row = tile * 16 + l15;     // A-matrix row held by this lane
    const int col = wv * 16 + l15;       // B/C column held by this lane

    v8f c = {};
    for (int kk = 0; kk < DM; kk += 32) {
        v16h a, bm;
        #pragma unroll
        for (int v = 0; v < 8; ++v) {
            const int kb = kk + kpat(v, half);
            const float2 xv = *(const float2*)(x + (size_t)row * DM + kb);
            a[2 * v]     = (_Float16)xv.x;
            a[2 * v + 1] = (_Float16)xv.y;
            bm[2 * v]     = (_Float16)W[(size_t)kb * DH + col];
            bm[2 * v + 1] = (_Float16)W[(size_t)(kb + 1) * DH + col];
        }
        c = wmma_f16(a, bm, c);
    }

    const float bcol = bias[col];
    #pragma unroll
    for (int r = 0; r < 8; ++r) {
        const int g   = tile * 16 + r + 8 * half;   // global row (b*S + s)
        const float v = c[r] + bcol;
        if (P == 0) {
            dst[(size_t)g * DH + col] = (_Float16)(v * 0.125f);
        } else if (P == 1) {
            dst[(size_t)g * DH + col] = (_Float16)v;
        } else {
            const int b = g >> 12;              // / SEQ
            const int s = g & (SEQ - 1);
            dst[((size_t)(b * DH + col)) * SEQ + s] = (_Float16)v;
        }
    }
}

// ---------------------------------------------------------------------------
// Kernel B: flash attention.  block = 256 (8 waves), one 16-query tile per
// wave, online softmax over all SEQ keys in tiles of 64 (amortizes the
// shuffle reductions / rescale over 2x more keys; WMMA count unchanged).
// ---------------------------------------------------------------------------
__global__ __launch_bounds__(256) void flash_attn_kernel(
    const _Float16* __restrict__ Qh, const _Float16* __restrict__ Kh,
    const _Float16* __restrict__ Vt, const int* __restrict__ mask,
    float* __restrict__ out)
{
    __shared__ float lds[8][16 * 65];    // per-wave 16x64 P staging (+pad)

    const int lane = threadIdx.x & 31;
    const int wid  = threadIdx.x >> 5;
    const int half = lane >> 4;
    const int l15  = lane & 15;

    const int qt  = blockIdx.x * 8 + wid;
    const int b   = qt / (SEQ / 16);
    const int qr0 = (qt % (SEQ / 16)) * 16;

    float* my = &lds[wid][0];

    // Q A-tiles (held for the whole key loop): rows qr0..qr0+15, d = 0..63.
    const _Float16* qp = Qh + ((size_t)(b * SEQ + qr0 + l15)) * DH;
    u32v8 uq0, uq1;
    #pragma unroll
    for (int v = 0; v < 8; ++v) {
        const int d = kpat(v, half);
        uq0[v] = *(const unsigned int*)(qp + d);
        uq1[v] = *(const unsigned int*)(qp + 32 + d);
    }
    const v16h aq0 = __builtin_bit_cast(v16h, uq0);
    const v16h aq1 = __builtin_bit_cast(v16h, uq1);

    float m[8], l[8];
    v8f acc[4] = {};
    #pragma unroll
    for (int r = 0; r < 8; ++r) { m[r] = -__builtin_inff(); l[r] = 0.f; }

    for (int kt = 0; kt < SEQ; kt += 64) {
        // ---- scores: four 16x16 tiles, K-dim = 64 (two WMMAs each) ----
        v8f c[4];
        int mk[4];
        #pragma unroll
        for (int j = 0; j < 4; ++j) {
            const int key = kt + 16 * j + l15;
            const _Float16* kp = Kh + ((size_t)(b * SEQ + key)) * DH;
            u32v8 u0, u1;
            #pragma unroll
            for (int v = 0; v < 8; ++v) {
                const int d = kpat(v, half);
                u0[v] = *(const unsigned int*)(kp + d);
                u1[v] = *(const unsigned int*)(kp + 32 + d);
            }
            v8f cc = {};
            cc = wmma_f16(aq0, __builtin_bit_cast(v16h, u0), cc);
            cc = wmma_f16(aq1, __builtin_bit_cast(v16h, u1), cc);
            c[j]  = cc;
            mk[j] = mask[b * SEQ + key];
        }

        // ---- online softmax over this 64-key tile ----
        #pragma unroll
        for (int r = 0; r < 8; ++r) {
            float loc = fmaxf(fmaxf(c[0][r], c[1][r]),
                              fmaxf(c[2][r], c[3][r]));
            #pragma unroll
            for (int off = 1; off < 16; off <<= 1)
                loc = fmaxf(loc, __shfl_xor(loc, off, 32));
            const float newm = fmaxf(m[r], loc);
            const float sc   = __expf(m[r] - newm);
            float p[4], rs = 0.f;
            #pragma unroll
            for (int j = 0; j < 4; ++j) {
                p[j] = __expf(c[j][r] - newm);
                rs  += p[j];
            }
            #pragma unroll
            for (int off = 1; off < 16; off <<= 1)
                rs += __shfl_xor(rs, off, 32);
            l[r] = l[r] * sc + rs;   // denominator includes masked keys
            m[r] = newm;
            #pragma unroll
            for (int t = 0; t < 4; ++t) acc[t][r] *= sc;
            // stage P in C-layout; mask applied AFTER softmax (-inf fill)
            const int rowL = r + 8 * half;
            #pragma unroll
            for (int j = 0; j < 4; ++j)
                my[rowL * 65 + 16 * j + l15] =
                    mk[j] ? p[j] : -__builtin_inff();
        }

        // ---- re-read P in A-layout: two 16x32 f16 tiles ----
        v16h pa0, pa1;
        #pragma unroll
        for (int v = 0; v < 8; ++v) {
            const int k = kpat(v, half);
            pa0[2 * v]     = (_Float16)my[l15 * 65 + k];
            pa0[2 * v + 1] = (_Float16)my[l15 * 65 + k + 1];
            pa1[2 * v]     = (_Float16)my[l15 * 65 + 32 + k];
            pa1[2 * v + 1] = (_Float16)my[l15 * 65 + 32 + k + 1];
        }

        // ---- acc += P(16x64) x V(64x16) for 4 output-dim tiles ----
        #pragma unroll
        for (int t = 0; t < 4; ++t) {
            const _Float16* vp =
                Vt + ((size_t)(b * DH + t * 16 + l15)) * SEQ + kt;
            u32v8 uv0, uv1;
            #pragma unroll
            for (int v = 0; v < 8; ++v) {
                const int k = kpat(v, half);
                uv0[v] = *(const unsigned int*)(vp + k);
                uv1[v] = *(const unsigned int*)(vp + 32 + k);
            }
            acc[t] = wmma_f16(pa0, __builtin_bit_cast(v16h, uv0), acc[t]);
            acc[t] = wmma_f16(pa1, __builtin_bit_cast(v16h, uv1), acc[t]);
        }
    }

    // ---- normalize and store (B, S, 64) fp32 ----
    #pragma unroll
    for (int r = 0; r < 8; ++r) {
        const float invl = 1.0f / l[r];
        const int   M    = r + 8 * half;
        #pragma unroll
        for (int t = 0; t < 4; ++t) {
            const int dd = t * 16 + l15;
            out[((size_t)(b * SEQ + qr0 + M)) * DH + dd] = acc[t][r] * invl;
        }
    }
}

// ---------------------------------------------------------------------------
extern "C" void kernel_launch(void* const* d_in, const int* in_sizes, int n_in,
                              void* d_out, int out_size, void* d_ws,
                              size_t ws_size, hipStream_t stream)
{
    const float* x    = (const float*)d_in[0];
    const int*   mask = (const int*)d_in[1];
    const float* Wq   = (const float*)d_in[2];
    const float* bq   = (const float*)d_in[3];
    const float* Wk   = (const float*)d_in[4];
    const float* bk   = (const float*)d_in[5];
    const float* Wv   = (const float*)d_in[6];
    const float* bv   = (const float*)d_in[7];
    float* out = (float*)d_out;

    const int BS = in_sizes[0] / DM;     // B * S rows

    _Float16* Qh = (_Float16*)d_ws;
    _Float16* Kh = Qh + (size_t)BS * DH;
    _Float16* Vt = Kh + (size_t)BS * DH;

    const int tiles = BS / 16;
    qkv_proj_kernel<0><<<tiles, 128, 0, stream>>>(x, Wq, bq, Qh);
    qkv_proj_kernel<1><<<tiles, 128, 0, stream>>>(x, Wk, bk, Kh);
    qkv_proj_kernel<2><<<tiles, 128, 0, stream>>>(x, Wv, bv, Vt);

    // one 16-query tile per wave, 8 waves per block
    flash_attn_kernel<<<tiles / 8, 256, 0, stream>>>(Qh, Kh, Vt, mask, out);
}
